// EllipticGNN_52518860095815
// MI455X (gfx1250) — compile-verified, matrix operand order checked by reading, not run on hardware
//
#include <hip/hip_runtime.h>

#define FH 128        // hidden width
#define FIN 165       // input features
#define KPAD1 192     // 165 padded to multiple of 32
#define NOUT 2

typedef __attribute__((ext_vector_type(16))) __bf16 v16bf;
typedef __attribute__((ext_vector_type(8)))  __bf16 v8bf;
typedef __attribute__((ext_vector_type(8)))  float  v8f;

// round-to-nearest-even f32 -> bf16 via bit tricks
__device__ __forceinline__ __bf16 f2bf(float f) {
    union { float f; unsigned u; } v; v.f = f;
    unsigned r = v.u + 0x7FFFu + ((v.u >> 16) & 1u);
    unsigned short h = (unsigned short)(r >> 16);
    return __builtin_bit_cast(__bf16, h);
}

// ---------------- degree / norm ----------------
__global__ void deg_kernel(const int* __restrict__ dst, float* __restrict__ deg, int E) {
    int t = blockIdx.x * blockDim.x + threadIdx.x;
    if (t < E) atomicAdd(&deg[dst[t]], 1.0f);
}

__global__ void rsqrt_kernel(float* __restrict__ dinv, int N) {
    int t = blockIdx.x * blockDim.x + threadIdx.x;
    if (t < N) dinv[t] = rsqrtf(dinv[t] + 1.0f);   // deg + self-loop
}

// ---------------- pack x: f32 [N x FIN] -> bf16 [Npad x KPAD1], zero-padded ----------
__global__ void packA_kernel(const float* __restrict__ X, __bf16* __restrict__ Xb,
                             int N, int Npad, int K, int Kpad) {
    int t = blockIdx.x * blockDim.x + threadIdx.x;
    if (t >= Npad * Kpad) return;
    int row = t / Kpad, k = t - row * Kpad;
    float v = (row < N && k < K) ? X[(size_t)row * K + k] : 0.0f;
    Xb[t] = f2bf(v);
}

// ---------------- pack W into WMMA B-fragment order ----------------
// Bf[(kblk*8 + ct)*512 + lane*16 + j]; lane: col = ct*16 + (lane&15),
// K half-select ksel = (lane&16)?8:0; j<8 -> k=kblk*32+ksel+j, j>=8 -> +8 more.
__global__ void packB_kernel(const float* __restrict__ W, __bf16* __restrict__ Bf,
                             int K, int Kpad) {
    int t = blockIdx.x * blockDim.x + threadIdx.x;
    if (t >= Kpad * FH) return;
    int kblk = t >> 12;
    int r    = t & 4095;
    int ct   = r >> 9;
    int r2   = r & 511;
    int lane = r2 >> 4;
    int j    = r2 & 15;
    int ksel = (lane & 16) ? 8 : 0;
    int k    = kblk * 32 + ksel + ((j < 8) ? j : j + 8);
    int col  = ct * 16 + (lane & 15);
    float v  = (k < K) ? W[(size_t)k * FH + col] : 0.0f;
    Bf[t] = f2bf(v);
}

// ---------------- WMMA GEMM: C[N x 128] = A[Npad x KPAD](bf16) * W(frag bf16) --------
// One wave per 16-row stripe computes the full 16x128 strip (8 C tiles).
// A fragment loaded once per K-step, reused across 8 WMMAs.
template<int KPAD>
__global__ void gcn_gemm_wmma(const __bf16* __restrict__ A,
                              const __bf16* __restrict__ Bf,
                              float* __restrict__ C, int N, int Npad) {
    const int lane    = threadIdx.x & 31;
    const int wave    = threadIdx.x >> 5;
    const int stripe  = blockIdx.x * 8 + wave;
    const int rowbase = stripe * 16;
    if (rowbase >= Npad) return;                  // uniform per wave: EXEC stays all-ones
    const int hl   = lane & 15;
    const int ksel = (lane & 16) ? 8 : 0;

    const __bf16* aptr = A + (size_t)(rowbase + hl) * KPAD + ksel;

    v8f acc[8];
#pragma unroll
    for (int i = 0; i < 8; ++i) acc[i] = (v8f){};

#pragma unroll
    for (int kb = 0; kb < KPAD; kb += 32) {
        union { v16bf v; v8bf h[2]; } a;
        a.h[0] = *(const v8bf*)(aptr + kb);        // K klo .. klo+7
        a.h[1] = *(const v8bf*)(aptr + kb + 16);   // K klo+16 .. klo+23
        const __bf16* bbase = Bf + (size_t)(kb >> 5) * 4096 + lane * 16;
#pragma unroll
        for (int ct = 0; ct < 8; ++ct) {
            union { v16bf v; v8bf h[2]; } b;
            b.h[0] = *(const v8bf*)(bbase + ct * 512);
            b.h[1] = *(const v8bf*)(bbase + ct * 512 + 8);
            acc[ct] = __builtin_amdgcn_wmma_f32_16x16x32_bf16(
                          false, a.v, false, b.v, (short)0, acc[ct], false, false);
        }
    }
    // C/D layout: VGPR r, lane l -> M = r + 8*(l>=16), N = l&15
    const int mbase = rowbase + ((lane & 16) ? 8 : 0);
#pragma unroll
    for (int ct = 0; ct < 8; ++ct) {
#pragma unroll
        for (int r = 0; r < 8; ++r) {
            int m = mbase + r;
            if (m < N) C[(size_t)m * FH + ct * 16 + hl] = acc[ct][r];
        }
    }
}

// ---------------- edge aggregation: acc[dst] += h[src] * dinv[src]*dinv[dst] ----------
// One wave per edge; lane handles 4 contiguous features (32*4 = 128).
__global__ void aggregate_kernel(const float* __restrict__ h,
                                 const int* __restrict__ src, const int* __restrict__ dst,
                                 const float* __restrict__ dinv,
                                 float* __restrict__ acc, int E) {
    int t    = blockIdx.x * blockDim.x + threadIdx.x;
    int edge = t >> 5;
    int lane = t & 31;
    if (edge >= E) return;
    int s = src[edge], d = dst[edge];
    float w = dinv[s] * dinv[d];
    const float4* hrow = (const float4*)(h + (size_t)s * FH);
    float4 v = hrow[lane];
    float* arow = acc + (size_t)d * FH + lane * 4;
    atomicAdd(arow + 0, v.x * w);
    atomicAdd(arow + 1, v.y * w);
    atomicAdd(arow + 2, v.z * w);
    atomicAdd(arow + 3, v.w * w);
}

// ---------------- self-loop + bias + ReLU, emit bf16 (layer-1: feeds GEMM2) ----------
__global__ void selfrelu_bf16_kernel(const float* __restrict__ acc, const float* __restrict__ hlin,
                                     const float* __restrict__ dinv, const float* __restrict__ bias,
                                     __bf16* __restrict__ hb, int N, int Npad) {
    int t = blockIdx.x * blockDim.x + threadIdx.x;
    if (t >= Npad * FH) return;
    int i = t >> 7, f = t & 127;
    float v = 0.0f;
    if (i < N) {
        float di = dinv[i];
        v = acc[t] + hlin[t] * (di * di) + bias[f];
        v = v > 0.0f ? v : 0.0f;
    }
    hb[t] = f2bf(v);
}

// ---------------- self-loop + bias + ReLU, f32 in-place (layer-2: feeds classifier) ---
__global__ void selfrelu_f32_kernel(float* __restrict__ acc, const float* __restrict__ hlin,
                                    const float* __restrict__ dinv, const float* __restrict__ bias,
                                    int N) {
    int t = blockIdx.x * blockDim.x + threadIdx.x;
    if (t >= N * FH) return;
    int i = t >> 7, f = t & 127;
    float di = dinv[i];
    float v = acc[t] + hlin[t] * (di * di) + bias[f];
    acc[t] = v > 0.0f ? v : 0.0f;
}

// ---------------- classifier: out = h @ W3 + b3 (128 -> 2) ----------------
__global__ void final_kernel(const float* __restrict__ h, const float* __restrict__ W3,
                             const float* __restrict__ b3, float* __restrict__ out, int N) {
    int i = blockIdx.x * blockDim.x + threadIdx.x;
    if (i >= N) return;
    const float* hr = h + (size_t)i * FH;
    float s0 = 0.0f, s1 = 0.0f;
#pragma unroll 8
    for (int k = 0; k < FH; ++k) {
        float v = hr[k];
        s0 += v * W3[k * NOUT + 0];
        s1 += v * W3[k * NOUT + 1];
    }
    out[(size_t)i * NOUT + 0] = s0 + b3[0];
    out[(size_t)i * NOUT + 1] = s1 + b3[1];
}

static inline size_t align256(size_t x) { return (x + 255) & ~(size_t)255; }

extern "C" void kernel_launch(void* const* d_in, const int* in_sizes, int n_in,
                              void* d_out, int out_size, void* d_ws, size_t ws_size,
                              hipStream_t stream) {
    const float* x  = (const float*)d_in[0];
    const int*   ei = (const int*)d_in[1];
    const float* W1 = (const float*)d_in[2];
    const float* b1 = (const float*)d_in[3];
    const float* W2 = (const float*)d_in[4];
    const float* b2 = (const float*)d_in[5];
    const float* W3 = (const float*)d_in[6];
    const float* b3 = (const float*)d_in[7];

    const int N    = in_sizes[0] / FIN;
    const int E    = in_sizes[1] / 2;
    const int Npad = (N + 15) & ~15;
    const int* src = ei;
    const int* dst = ei + E;

    // workspace carve-up
    char* ws = (char*)d_ws;
    size_t off = 0;
    float*  dinv = (float*)(ws + off);  off = align256(off + (size_t)N * 4);
    float*  hlin = (float*)(ws + off);  off = align256(off + (size_t)N * FH * 4);
    float*  hacc = (float*)(ws + off);  off = align256(off + (size_t)N * FH * 4);
    __bf16* Ab   = (__bf16*)(ws + off); off = align256(off + (size_t)Npad * KPAD1 * 2); // xb; later reused as h1(bf16)
    __bf16* Bf   = (__bf16*)(ws + off); off = align256(off + (size_t)KPAD1 * FH * 2);

    const int T = 256;
    const int nStripes    = Npad / 16;
    const int nGemmBlocks = (nStripes + 7) / 8;               // 8 waves per block
    const int nElemBlocks = ((int)((size_t)N * FH) + T - 1) / T;
    const int nPadBlocks  = ((int)((size_t)Npad * FH) + T - 1) / T;
    const int nEdgeBlocks = (int)(((long long)E * 32 + T - 1) / T);

    // degrees -> dinv
    hipMemsetAsync(dinv, 0, (size_t)N * 4, stream);
    deg_kernel<<<(E + T - 1) / T, T, 0, stream>>>(dst, dinv, E);
    rsqrt_kernel<<<(N + T - 1) / T, T, 0, stream>>>(dinv, N);

    // ---- layer 1 ----
    packA_kernel<<<((Npad * KPAD1) + T - 1) / T, T, 0, stream>>>(x, Ab, N, Npad, FIN, KPAD1);
    packB_kernel<<<(KPAD1 * FH + T - 1) / T, T, 0, stream>>>(W1, Bf, FIN, KPAD1);
    gcn_gemm_wmma<KPAD1><<<nGemmBlocks, T, 0, stream>>>(Ab, Bf, hlin, N, Npad);
    hipMemsetAsync(hacc, 0, (size_t)N * FH * 4, stream);
    aggregate_kernel<<<nEdgeBlocks, T, 0, stream>>>(hlin, src, dst, dinv, hacc, E);
    // h1 as bf16, overwriting the x-pack region (x no longer needed)
    selfrelu_bf16_kernel<<<nPadBlocks, T, 0, stream>>>(hacc, hlin, dinv, b1, Ab, N, Npad);

    // ---- layer 2 ----
    packB_kernel<<<(FH * FH + T - 1) / T, T, 0, stream>>>(W2, Bf, FH, FH);
    gcn_gemm_wmma<FH><<<nGemmBlocks, T, 0, stream>>>(Ab, Bf, hlin, N, Npad);
    hipMemsetAsync(hacc, 0, (size_t)N * FH * 4, stream);
    aggregate_kernel<<<nEdgeBlocks, T, 0, stream>>>(hlin, src, dst, dinv, hacc, E);
    selfrelu_f32_kernel<<<nElemBlocks, T, 0, stream>>>(hacc, hlin, dinv, b2, N);   // hacc = h2

    // ---- classifier ----
    final_kernel<<<(N + T - 1) / T, T, 0, stream>>>(hacc, W3, b3, (float*)d_out, N);
}
